// CrossAttention_30219389894968
// MI455X (gfx1250) — matrix operand order repaired and checked by reading, not compiled, not run on hardware
//
#include <hip/hip_runtime.h>
#include <hip/hip_bf16.h>

// ---------------------------------------------------------------------------
// Cross-attention for MI455X (gfx1250, wave32, WMMA).
// Pipeline:
//   1) convert x/context/Wq/Wk/Wv/Wo to f16 in workspace
//   2) Q = x@Wq^T   -> [B,H,Nq,64]   (f16, per-head)
//      K = c@Wk^T   -> [B,H,Nk,64]   (f16, per-head)
//      V = c@Wv^T   -> [B,H,64,Nk]   (f16, per-head, TRANSPOSED so that the
//                                     P@V B-operand is a contiguous [n][k] load)
//   3) flash attention (online softmax, bias added, scale = 1/sqrt(64))
//      -> Attn [B*Nq, 512] f16
//   4) out = Attn@Wo^T + bo -> f32 d_out
// All matmuls use v_wmma_f32_16x16x32_f16. Next-tile global_prefetch_b8
// hides HBM latency behind the WMMA stream.
// ---------------------------------------------------------------------------

typedef __attribute__((ext_vector_type(16))) _Float16 v16h;
typedef __attribute__((ext_vector_type(8)))  float    v8f;
typedef __attribute__((ext_vector_type(4)))  _Float16 v4h;

#define WMMA_F16(a, b, c) \
  __builtin_amdgcn_wmma_f32_16x16x32_f16(false, (a), false, (b), (short)0, (c), false, false)

// A-fragment (16x32 f16, MxK). base points at row 0 / K 0; stride in elements.
// ISA layout: lanes 0-15 hold row r, K{0..7, 16..23}; lanes 16-31 K{8..15, 24..31}.
__device__ inline v16h load_frag_a(const _Float16* base, int stride, int lane) {
  int r = lane & 15, h = (lane >> 4) & 1;
  const _Float16* p = base + (size_t)r * stride + h * 8;
  v16h f;
  ((uint4*)&f)[0] = *(const uint4*)(p);        // K off+0 .. off+7
  ((uint4*)&f)[1] = *(const uint4*)(p + 16);   // K off+16 .. off+23
  return f;
}

// B-fragment (32x16 f16, KxN) supplied as B^T stored row-major [n][k].
// lanes 0-15: col n = lane, K{0..7, 16..23} ; lanes 16-31: col n = lane-16, K{8..15, 24..31}.
__device__ inline v16h load_frag_b(const _Float16* baseT, int stride, int lane) {
  int r = lane & 15, h = (lane >> 4) & 1;
  const _Float16* p = baseT + (size_t)r * stride + h * 8;
  v16h f;
  ((uint4*)&f)[0] = *(const uint4*)(p);        // K h*8+0 .. +7
  ((uint4*)&f)[1] = *(const uint4*)(p + 16);   // K 16+h*8+0 .. +7
  return f;
}

// ---------------------------------------------------------------------------
// f32 -> f16 conversion, 4 elements / thread
// ---------------------------------------------------------------------------
__global__ void cvt_f32_to_f16(const float* __restrict__ in,
                               _Float16* __restrict__ out, int n4) {
  int i = blockIdx.x * blockDim.x + threadIdx.x;
  if (i < n4) {
    float4 v = ((const float4*)in)[i];
    v4h o = {(_Float16)v.x, (_Float16)v.y, (_Float16)v.z, (_Float16)v.w};
    ((v4h*)out)[i] = o;
  }
}

// ---------------------------------------------------------------------------
// GEMM: C[M x 512] = A[M x 512] * W^T,  W given row-major [512 x 512] ([n][k]).
// Block tile 128(M) x 128(N), K-step 32, 8 waves in a 4(M) x 2(N) grid;
// each wave computes a 32x64 tile (8 WMMAs / k-step, B-fragments reused x2).
// MODE 1: f16 out, per-head [B,H,Nseq,64]            (Q, K)
// MODE 2: f16 out, per-head transposed [B,H,64,Nseq] (V^T)
// MODE 3: f32 out + bias vector, row-major [M x 512] (final projection)
// ---------------------------------------------------------------------------
template <int MODE>
__global__ __launch_bounds__(256) void gemm512_wmma(
    const _Float16* __restrict__ A, const _Float16* __restrict__ Bt,
    void* __restrict__ dstv, int qshift, const float* __restrict__ bo) {
  constexpr int KD = 512, ND = 512;
  constexpr int LDT = 32 + 8;                 // pad keeps 16B alignment
  __shared__ __align__(16) _Float16 sA[128 * LDT];
  __shared__ __align__(16) _Float16 sB[128 * LDT];

  const int tid = threadIdx.x;
  const int lane = tid & 31, wave = tid >> 5;
  const int r = lane & 15, hlf = lane >> 4;
  const int wm = wave & 3;                    // wave M position (32 rows)
  const int wn = wave >> 2;                   // wave N position (64 cols)
  const int m0 = blockIdx.x * 128;
  const int n0 = blockIdx.y * 128;

  const _Float16* Ablk = A + (size_t)m0 * KD;
  const _Float16* Bblk = Bt + (size_t)n0 * KD;

  v8f acc[2][4] = {};

  const int row = tid >> 1, colh = (tid & 1) * 16;  // cooperative tile loads

  for (int ks = 0; ks < KD; ks += 32) {
    __syncthreads();
    {
      const uint4* ga = (const uint4*)(Ablk + (size_t)row * KD + ks + colh);
      uint4* da = (uint4*)(sA + row * LDT + colh);
      da[0] = ga[0]; da[1] = ga[1];
      const uint4* gb = (const uint4*)(Bblk + (size_t)row * KD + ks + colh);
      uint4* db = (uint4*)(sB + row * LDT + colh);
      db[0] = gb[0]; db[1] = gb[1];
    }
    // pull the next K-slab toward the caches while this one is consumed
    if (ks + 32 < KD) {
      __builtin_prefetch(Ablk + (size_t)row * KD + ks + 32 + colh, 0, 1);
      __builtin_prefetch(Bblk + (size_t)row * KD + ks + 32 + colh, 0, 1);
    }
    __syncthreads();

    v16h a0 = load_frag_a(sA + (wm * 32 + 0) * LDT, LDT, lane);
    v16h a1 = load_frag_a(sA + (wm * 32 + 16) * LDT, LDT, lane);
#pragma unroll
    for (int j = 0; j < 4; ++j) {
      v16h b = load_frag_b(sB + (wn * 64 + j * 16) * LDT, LDT, lane);
      acc[0][j] = WMMA_F16(a0, b, acc[0][j]);
      acc[1][j] = WMMA_F16(a1, b, acc[1][j]);
    }
  }

  const int H = 8;
#pragma unroll
  for (int i = 0; i < 2; ++i) {
#pragma unroll
    for (int j = 0; j < 4; ++j) {
      int n = n0 + wn * 64 + j * 16 + r;
#pragma unroll
      for (int e = 0; e < 8; ++e) {
        int m = m0 + wm * 32 + i * 16 + 8 * hlf + e;
        float val = acc[i][j][e];
        if (MODE == 3) {
          ((float*)dstv)[(size_t)m * ND + n] = val + bo[n];
        } else {
          int b = m >> qshift, q = m & ((1 << qshift) - 1);
          int h = n >> 6, d = n & 63;
          int nseq = 1 << qshift;
          _Float16 hv = (_Float16)val;
          if (MODE == 1)
            ((_Float16*)dstv)[(((size_t)(b * H + h)) * nseq + q) * 64 + d] = hv;
          else  // MODE == 2 : transposed (V^T)
            ((_Float16*)dstv)[(((size_t)(b * H + h)) * 64 + d) * nseq + q] = hv;
        }
      }
    }
  }
}

// ---------------------------------------------------------------------------
// Flash attention. 128 threads = 4 waves; each wave owns 16 query rows.
// grid = (Nq/64, B*H). K/V staged in LDS in 64-key chunks.
// ---------------------------------------------------------------------------
__global__ __launch_bounds__(128) void flash_attn_wmma(
    const _Float16* __restrict__ Qh, const _Float16* __restrict__ Kh,
    const _Float16* __restrict__ Vth, const float* __restrict__ bias,
    _Float16* __restrict__ Oh) {
  constexpr int NQ = 1024, NK = 2048, DH = 64, H = 8;
  constexpr float SCALE = 0.125f;  // 64^-0.5
  constexpr int LDK = DH + 8, LDV = DH + 8;
  __shared__ __align__(16) _Float16 sK[64 * LDK];
  __shared__ __align__(16) _Float16 sV[64 * LDV];
  __shared__ __align__(16) _Float16 sP[4][16 * 64];

  const int tid = threadIdx.x;
  const int lane = tid & 31, wave = tid >> 5;
  const int r = lane & 15, hlf = lane >> 4;
  const int bh = blockIdx.y;          // b*H + h
  const int b = bh >> 3, h = bh & 7;
  const int q0 = blockIdx.x * 64 + wave * 16;

  // Q fragments live in registers for the whole Nk sweep
  const _Float16* Qbase = Qh + (((size_t)bh) * NQ + q0) * DH;
  v16h qf[2];
#pragma unroll
  for (int kk = 0; kk < 2; ++kk) qf[kk] = load_frag_a(Qbase + kk * 32, DH, lane);

  v8f o[4] = {};
  float m[8], ssum[8];
#pragma unroll
  for (int e = 0; e < 8; ++e) { m[e] = -__builtin_inff(); ssum[e] = 0.f; }

  const float* biasBase = bias + ((size_t)b * NQ + q0) * NK;
  const _Float16* Kbase = Kh + (size_t)bh * NK * DH;
  const _Float16* Vbase = Vth + (size_t)bh * DH * NK;

  const int lrow = tid >> 1, lcol = (tid & 1) * 32;  // cooperative tile load

  for (int kc = 0; kc < NK; kc += 64) {
    __syncthreads();
    {
      const uint4* sk = (const uint4*)(Kbase + ((size_t)(kc + lrow)) * DH + lcol);
      uint4* dk = (uint4*)(sK + lrow * LDK + lcol);
      dk[0] = sk[0]; dk[1] = sk[1]; dk[2] = sk[2]; dk[3] = sk[3];
      const uint4* sv = (const uint4*)(Vbase + (size_t)lrow * NK + kc + lcol);
      uint4* dv = (uint4*)(sV + lrow * LDV + lcol);
      dv[0] = sv[0]; dv[1] = sv[1]; dv[2] = sv[2]; dv[3] = sv[3];
    }
    // prefetch next chunk's K/V while this chunk is computed
    if (kc + 64 < NK) {
      __builtin_prefetch(Kbase + ((size_t)(kc + 64 + lrow)) * DH + lcol, 0, 1);
      __builtin_prefetch(Vbase + (size_t)lrow * NK + (kc + 64) + lcol, 0, 1);
    }
    __syncthreads();

    // S(16x64) = Q(16x64) @ K^T  -- sK is [key][d] == B^T layout [n][k]
    v8f s[4];
#pragma unroll
    for (int j = 0; j < 4; ++j) {
      v8f sj = {};
      sj = WMMA_F16(qf[0], load_frag_b(sK + j * 16 * LDK + 0, LDK, lane), sj);
      sj = WMMA_F16(qf[1], load_frag_b(sK + j * 16 * LDK + 32, LDK, lane), sj);
      s[j] = sj;
    }

    // scale + bias
#pragma unroll
    for (int j = 0; j < 4; ++j)
#pragma unroll
      for (int e = 0; e < 8; ++e) {
        float bl = biasBase[((size_t)(e + 8 * hlf)) * NK + kc + j * 16 + r];
        s[j][e] = s[j][e] * SCALE + bl;
      }

    // online softmax: row stats replicated across the 16 N-lanes of each half
#pragma unroll
    for (int e = 0; e < 8; ++e) {
      float v = fmaxf(fmaxf(s[0][e], s[1][e]), fmaxf(s[2][e], s[3][e]));
      v = fmaxf(v, __shfl_xor(v, 1, 32));
      v = fmaxf(v, __shfl_xor(v, 2, 32));
      v = fmaxf(v, __shfl_xor(v, 4, 32));
      v = fmaxf(v, __shfl_xor(v, 8, 32));
      float mnew = fmaxf(m[e], v);
      float alpha = __expf(m[e] - mnew);
      float rs = 0.f;
#pragma unroll
      for (int j = 0; j < 4; ++j) {
        float p = __expf(s[j][e] - mnew);
        s[j][e] = p;
        rs += p;
      }
      rs += __shfl_xor(rs, 1, 32);
      rs += __shfl_xor(rs, 2, 32);
      rs += __shfl_xor(rs, 4, 32);
      rs += __shfl_xor(rs, 8, 32);
      ssum[e] = ssum[e] * alpha + rs;
      m[e] = mnew;
#pragma unroll
      for (int j = 0; j < 4; ++j) o[j][e] *= alpha;
    }

    // P -> wave-private LDS (D-layout -> A-layout shuffle through LDS)
#pragma unroll
    for (int j = 0; j < 4; ++j)
#pragma unroll
      for (int e = 0; e < 8; ++e)
        sP[wave][(e + 8 * hlf) * 64 + j * 16 + r] = (_Float16)s[j][e];

    // O(16x64) += P(16x64) @ V(64x64) -- sV is [d][key] == B^T layout [n][k]
#pragma unroll
    for (int kk = 0; kk < 2; ++kk) {
      v16h pa = load_frag_a(&sP[wave][0] + kk * 32, 64, lane);
#pragma unroll
      for (int j = 0; j < 4; ++j) {
        v16h bv = load_frag_b(sV + j * 16 * LDV + kk * 32, LDV, lane);
        o[j] = WMMA_F16(pa, bv, o[j]);
      }
    }
  }

  // normalize and write Attn [B*Nq, 512] (col = h*64 + d)
#pragma unroll
  for (int j = 0; j < 4; ++j)
#pragma unroll
    for (int e = 0; e < 8; ++e) {
      int q = q0 + e + 8 * hlf;
      int d = h * 64 + j * 16 + r;
      float val = o[j][e] / ssum[e];
      Oh[((size_t)(b * NQ + q)) * 512 + d] = (_Float16)val;
    }
}

// ---------------------------------------------------------------------------
extern "C" void kernel_launch(void* const* d_in, const int* in_sizes, int n_in,
                              void* d_out, int out_size, void* d_ws, size_t ws_size,
                              hipStream_t stream) {
  (void)in_sizes; (void)n_in; (void)out_size; (void)ws_size;
  const float* x    = (const float*)d_in[0];  // [4,1024,512]
  const float* ctx  = (const float*)d_in[1];  // [4,2048,512]
  const float* bias = (const float*)d_in[2];  // [4,1024,2048]
  const float* Wq   = (const float*)d_in[3];  // [512,512]
  const float* Wk   = (const float*)d_in[4];
  const float* Wv   = (const float*)d_in[5];
  const float* Wo   = (const float*)d_in[6];
  const float* bo   = (const float*)d_in[7];  // [512]

  constexpr size_t NX = (size_t)4 * 1024 * 512;   // x elems
  constexpr size_t NC = (size_t)4 * 2048 * 512;   // context elems
  constexpr size_t NW = (size_t)512 * 512;        // weight elems

  // workspace carve (f16): ~38 MB total
  char* ws = (char*)d_ws;
  _Float16* xh  = (_Float16*)ws;  ws += NX * 2;
  _Float16* ch  = (_Float16*)ws;  ws += NC * 2;
  _Float16* Wqh = (_Float16*)ws;  ws += NW * 2;
  _Float16* Wkh = (_Float16*)ws;  ws += NW * 2;
  _Float16* Wvh = (_Float16*)ws;  ws += NW * 2;
  _Float16* Woh = (_Float16*)ws;  ws += NW * 2;
  _Float16* Qh  = (_Float16*)ws;  ws += NX * 2;   // [B,H,1024,64]
  _Float16* Kh  = (_Float16*)ws;  ws += NC * 2;   // [B,H,2048,64]
  _Float16* Vth = (_Float16*)ws;  ws += NC * 2;   // [B,H,64,2048]
  _Float16* Ah  = (_Float16*)ws;  ws += NX * 2;   // [4096,512]

  // 1) conversions
  cvt_f32_to_f16<<<(NX / 4 + 255) / 256, 256, 0, stream>>>(x, xh, (int)(NX / 4));
  cvt_f32_to_f16<<<(NC / 4 + 255) / 256, 256, 0, stream>>>(ctx, ch, (int)(NC / 4));
  cvt_f32_to_f16<<<(NW / 4 + 255) / 256, 256, 0, stream>>>(Wq, Wqh, (int)(NW / 4));
  cvt_f32_to_f16<<<(NW / 4 + 255) / 256, 256, 0, stream>>>(Wk, Wkh, (int)(NW / 4));
  cvt_f32_to_f16<<<(NW / 4 + 255) / 256, 256, 0, stream>>>(Wv, Wvh, (int)(NW / 4));
  cvt_f32_to_f16<<<(NW / 4 + 255) / 256, 256, 0, stream>>>(Wo, Woh, (int)(NW / 4));

  // 2) projections (M=4096 -> qshift 10, M=8192 -> qshift 11), N blocks = 512/128
  gemm512_wmma<1><<<dim3(32, 4), 256, 0, stream>>>(xh, Wqh, Qh, 10, nullptr);
  gemm512_wmma<1><<<dim3(64, 4), 256, 0, stream>>>(ch, Wkh, Kh, 11, nullptr);
  gemm512_wmma<2><<<dim3(64, 4), 256, 0, stream>>>(ch, Wvh, Vth, 11, nullptr);

  // 3) flash attention: grid (Nq/64, B*H)
  flash_attn_wmma<<<dim3(16, 32), 128, 0, stream>>>(Qh, Kh, Vth, bias, Ah);

  // 4) output projection + bias -> f32 d_out
  gemm512_wmma<3><<<dim3(32, 4), 256, 0, stream>>>(Ah, Woh, d_out, 10, bo);
}
